// QGNN_87926570483846
// MI455X (gfx1250) — matrix operand order repaired and checked by reading, not compiled, not run on hardware
//
#include <hip/hip_runtime.h>

// ---------------------------------------------------------------------------
// CDNA5 (gfx1250) QGNN forward.
// Strategy: bf16 activations/weights + V_WMMA_F32_16X16X32_BF16, fp32
// accumulation + native fp32 global atomics for the segment sums, L2
// normalize in fp32. One wave = 16 rows x 128 cols (8 WMMA accumulator
// tiles). Weights staged global->LDS via GLOBAL_LOAD_ASYNC_TO_LDS_B128
// (ASYNCcnt) — builtin confirmed present in this toolchain.
// ---------------------------------------------------------------------------

typedef __attribute__((ext_vector_type(16))) __bf16 v16bf;
typedef __attribute__((ext_vector_type(8)))  float  v8f;

// Matches the builtin's parameter type: 'int __vector(4)' pointers in
// addrspace(1) (printed as __device__) / addrspace(3) (__shared__).
typedef int qint4 __attribute__((vector_size(16)));
#define QG_AS1 __attribute__((address_space(1)))
#define QG_AS3 __attribute__((address_space(3)))

#if defined(__has_builtin)
#if __has_builtin(__builtin_amdgcn_global_load_async_to_lds_b128)
#define QG_ASYNC_LDS 1
#endif
#if __has_builtin(__builtin_amdgcn_s_wait_asynccnt)
#define QG_WAIT_ASYNC() __builtin_amdgcn_s_wait_asynccnt(0)
#endif
#endif
#ifndef QG_ASYNC_LDS
#define QG_ASYNC_LDS 0
#endif
#ifndef QG_WAIT_ASYNC
#define QG_WAIT_ASYNC() asm volatile("s_wait_asynccnt 0x0" ::: "memory")
#endif

union Frag16 {
    v16bf v;
    uint4 q[2];
};

static __device__ __forceinline__ unsigned short f2bfbits(float f) {
    union { float f; unsigned u; } in;
    in.f = f;
    unsigned u = in.u;
    unsigned r = (u + 0x7FFFu + ((u >> 16) & 1u)) >> 16;   // round-to-nearest-even
    return (unsigned short)r;
}

static __device__ __forceinline__ __bf16 f2bf(float f) {
    union { unsigned short s; __bf16 b; } out;
    out.s = f2bfbits(f);
    return out.b;
}

// Native fp32 global atomic add (GLOBAL_ATOMIC_ADD_F32), avoiding CAS loops.
static __device__ __forceinline__ void atomAddF(float* p, float v) {
    unsafeAtomicAdd(p, v);
}

// Cooperative global->LDS weight staging for a 256-thread block.
// Async path: each lane issues a 16B GLOBAL_LOAD_ASYNC_TO_LDS_B128 (no VGPR
// bounce, tracked on ASYNCcnt), then waits ASYNCcnt==0 before the barrier.
static __device__ __forceinline__ void stage_weights(
    const unsigned short* __restrict__ g, unsigned short* s, int elems) {
    const uint4* src = (const uint4*)g;
    uint4*       dst = (uint4*)s;
    const int    n16 = elems >> 3;   // 8 shorts per uint4
#if QG_ASYNC_LDS
    for (int i = threadIdx.x; i < n16; i += 256) {
        __builtin_amdgcn_global_load_async_to_lds_b128(
            (QG_AS1 qint4*)(src + i),
            (QG_AS3 qint4*)(dst + i),
            /*offset=*/0, /*cpol=*/0);
    }
    QG_WAIT_ASYNC();
#else
    for (int i = threadIdx.x; i < n16; i += 256) dst[i] = src[i];
#endif
    __syncthreads();
}

// A-fragment (16x32 bf16, row = lane%16):
// elements 0..7  -> K = ks*32 + 8*(lane/16) + 0..7
// elements 8..15 -> K = ks*32 + 8*(lane/16) + 16..23
static __device__ __forceinline__ Frag16 load_a_bf16(const unsigned short* row,
                                                     int ks, int lane) {
    const int kb = ks * 32 + ((lane >> 4) << 3);
    Frag16 a;
    a.q[0] = *(const uint4*)(row + kb);
    a.q[1] = *(const uint4*)(row + kb + 16);
    return a;
}

// Same A layout, but source is fp32 (the h_N accumulator) -> convert to bf16.
static __device__ __forceinline__ Frag16 load_a_f32(const float* row,
                                                    int ks, int lane) {
    const int kb = ks * 32 + ((lane >> 4) << 3);
    float4 x0 = *(const float4*)(row + kb);
    float4 x1 = *(const float4*)(row + kb + 4);
    float4 y0 = *(const float4*)(row + kb + 16);
    float4 y1 = *(const float4*)(row + kb + 20);
    Frag16 a;
    a.v[0]  = f2bf(x0.x); a.v[1]  = f2bf(x0.y); a.v[2]  = f2bf(x0.z); a.v[3]  = f2bf(x0.w);
    a.v[4]  = f2bf(x1.x); a.v[5]  = f2bf(x1.y); a.v[6]  = f2bf(x1.z); a.v[7]  = f2bf(x1.w);
    a.v[8]  = f2bf(y0.x); a.v[9]  = f2bf(y0.y); a.v[10] = f2bf(y0.z); a.v[11] = f2bf(y0.w);
    a.v[12] = f2bf(y1.x); a.v[13] = f2bf(y1.y); a.v[14] = f2bf(y1.z); a.v[15] = f2bf(y1.w);
    return a;
}

// B-fragment (32x16 bf16) from LDS; weights stored column-major [128][KD]
// so each lane (column = tile*16 + lane%16) reads K values matching A:
// elements 0..7 -> K = ks*32 + 8*(lane/16) + 0..7, 8..15 -> + 16..23.
static __device__ __forceinline__ Frag16 load_b_lds(const unsigned short* sW,
                                                    int KD, int tile, int ks,
                                                    int lane) {
    const int n  = tile * 16 + (lane & 15);
    const int kb = ks * 32 + ((lane >> 4) << 3);
    const unsigned short* p = sW + (size_t)n * KD + kb;
    Frag16 b;
    b.q[0] = *(const uint4*)p;
    b.q[1] = *(const uint4*)(p + 16);
    return b;
}

// ---------------------------------------------------------------------------
// Utility kernels
// ---------------------------------------------------------------------------

__global__ void zero_f32(float* p, size_t n) {
    size_t i      = (size_t)blockIdx.x * blockDim.x + threadIdx.x;
    size_t stride = (size_t)gridDim.x * blockDim.x;
    for (; i < n; i += stride) p[i] = 0.0f;
}

// h0[i] = bf16(emb[gate_type[node]][j])
__global__ void embed_kernel(const int* __restrict__ gt,
                             const float* __restrict__ emb,
                             unsigned short* __restrict__ h0, int N) {
    size_t i = (size_t)blockIdx.x * blockDim.x + threadIdx.x;
    if (i >= (size_t)N * 64) return;
    int node = (int)(i >> 6);
    int j    = (int)(i & 63);
    h0[i] = f2bfbits(emb[(size_t)gt[node] * 64 + j]);
}

// dst[n*K + k] = bf16(src[k*128 + n])  (row-major [K][128] -> col-major [128][K])
__global__ void transpose_bf16_k(const float* __restrict__ src,
                                 unsigned short* __restrict__ dstp, int K) {
    int i = blockIdx.x * blockDim.x + threadIdx.x;
    if (i >= K * 128) return;
    int k = i % K;
    int n = i / K;
    dstp[(size_t)n * K + k] = f2bfbits(src[(size_t)k * 128 + n]);
}

// ---------------------------------------------------------------------------
// Edge message kernel: t = relu(h[src] @ W1_h + w @ W1_w + b1), then
// atomic scatter-add into hN[dst].  One wave: 16 edges x 128 outputs.
// ---------------------------------------------------------------------------

template <int DIN>
__global__ __launch_bounds__(256) void edge_msg_kernel(
    const unsigned short* __restrict__ hbf,   // [N][DIN] bf16
    const int* __restrict__ srcI, const int* __restrict__ dstI,
    const float* __restrict__ wE,             // [E][3]
    const float* __restrict__ W1,             // [(DIN+3)][128] fp32 (w-part rows)
    const float* __restrict__ b1,             // [128]
    const unsigned short* __restrict__ WT,    // [128][DIN] bf16 col-major
    float* __restrict__ hN, int E) {
    constexpr int KS = DIN / 32;
    __shared__ __align__(16) unsigned short sW[128 * DIN];
    stage_weights(WT, sW, 128 * DIN);

    const int lane  = threadIdx.x & 31;
    const int wave  = threadIdx.x >> 5;
    const int ebase = (blockIdx.x * 8 + wave) * 16;
    if (ebase >= E) return;   // wave-uniform exit (EXEC all-1s for WMMA)

    const int arow  = ebase + (lane & 15);
    const int aedge = (arow < E) ? arow : (E - 1);
    const unsigned short* ap = hbf + (size_t)srcI[aedge] * DIN;
    __builtin_prefetch(ap, 0, 0);

    v8f acc[8] = {};
#pragma unroll
    for (int ks = 0; ks < KS; ++ks) {
        Frag16 a = load_a_bf16(ap, ks, lane);
#pragma unroll
        for (int t = 0; t < 8; ++t) {
            Frag16 b = load_b_lds(sW, DIN, t, ks, lane);
            acc[t] = __builtin_amdgcn_wmma_f32_16x16x32_bf16(
                false, a.v, false, b.v, (short)0, acc[t], false, false);
        }
    }

    // Epilogue: + b1 + w-part, ReLU, scatter-add.
    // C layout: element r of acc[t] is row M = r + 8*(lane/16), col = t*16 + lane%16.
    const int half8 = (lane >> 4) << 3;
    const int n0    = lane & 15;
    float ew0[8], ew1[8], ew2[8];
    int   drow[8];
    bool  vld[8];
#pragma unroll
    for (int r = 0; r < 8; ++r) {
        int e  = ebase + half8 + r;
        vld[r] = (e < E);
        int ec = vld[r] ? e : (E - 1);
        ew0[r] = wE[(size_t)ec * 3 + 0];
        ew1[r] = wE[(size_t)ec * 3 + 1];
        ew2[r] = wE[(size_t)ec * 3 + 2];
        drow[r] = dstI[ec];
    }
#pragma unroll
    for (int t = 0; t < 8; ++t) {
        const int n    = t * 16 + n0;
        const float bb = b1[n];
        const float u0 = W1[(size_t)(DIN + 0) * 128 + n];
        const float u1 = W1[(size_t)(DIN + 1) * 128 + n];
        const float u2 = W1[(size_t)(DIN + 2) * 128 + n];
#pragma unroll
        for (int r = 0; r < 8; ++r) {
            if (!vld[r]) continue;
            float v = acc[t][r] + bb + ew0[r] * u0 + ew1[r] * u1 + ew2[r] * u2;
            v = fmaxf(v, 0.0f);
            atomAddF(&hN[(size_t)drow[r] * 128 + n], v);
        }
    }
}

// ---------------------------------------------------------------------------
// Node update: h = normalize(relu(h @ W2_top + hN @ W2_bot)); graph readout
// via atomics.  One wave: 16 nodes x 128 outputs.
// ---------------------------------------------------------------------------

template <int DIN>
__global__ __launch_bounds__(256) void node_update_kernel(
    const unsigned short* __restrict__ hbf,   // [N][DIN] bf16
    const float* __restrict__ hN,             // [N][128] fp32
    const unsigned short* __restrict__ WT2,   // [128][DIN+128] bf16 col-major
    const int* __restrict__ n2g,
    unsigned short* __restrict__ hout,        // [N][128] bf16
    float* __restrict__ hout32,               // nullable: final fp32 h
    float* __restrict__ readout,              // d_out readout base + layer*128
    int N) {
    constexpr int KD   = DIN + 128;
    constexpr int KS   = KD / 32;
    constexpr int KS_H = DIN / 32;
    __shared__ __align__(16) unsigned short sW[128 * KD];
    stage_weights(WT2, sW, 128 * KD);

    const int lane  = threadIdx.x & 31;
    const int wave  = threadIdx.x >> 5;
    const int nbase = (blockIdx.x * 8 + wave) * 16;
    if (nbase >= N) return;

    const int mrow = nbase + (lane & 15);
    const int mc   = (mrow < N) ? mrow : (N - 1);
    const unsigned short* ap = hbf + (size_t)mc * DIN;
    const float* fp          = hN + (size_t)mc * 128;
    __builtin_prefetch(fp, 0, 0);

    v8f acc[8] = {};
#pragma unroll
    for (int ks = 0; ks < KS; ++ks) {
        Frag16 a = (ks < KS_H) ? load_a_bf16(ap, ks, lane)
                               : load_a_f32(fp, ks - KS_H, lane);
#pragma unroll
        for (int t = 0; t < 8; ++t) {
            Frag16 b = load_b_lds(sW, KD, t, ks, lane);
            acc[t] = __builtin_amdgcn_wmma_f32_16x16x32_bf16(
                false, a.v, false, b.v, (short)0, acc[t], false, false);
        }
    }

    // ReLU + per-row sum of squares (row's 16 cols live in one 16-lane half).
    float ssq[8] = {0, 0, 0, 0, 0, 0, 0, 0};
#pragma unroll
    for (int t = 0; t < 8; ++t) {
#pragma unroll
        for (int r = 0; r < 8; ++r) {
            float v = fmaxf(acc[t][r], 0.0f);
            acc[t][r] = v;
            ssq[r] += v * v;
        }
    }
#pragma unroll
    for (int m = 1; m <= 8; m <<= 1) {
#pragma unroll
        for (int r = 0; r < 8; ++r) ssq[r] += __shfl_xor(ssq[r], m, 32);
    }

    const int half8 = (lane >> 4) << 3;
    const int n0    = lane & 15;
#pragma unroll
    for (int r = 0; r < 8; ++r) {
        int node = nbase + half8 + r;
        if (node >= N) continue;
        float sc = 1.0f / fmaxf(sqrtf(ssq[r]), 1e-12f);
        int g    = n2g[node];
#pragma unroll
        for (int t = 0; t < 8; ++t) {
            int n   = t * 16 + n0;
            float v = acc[t][r] * sc;
            hout[(size_t)node * 128 + n] = f2bfbits(v);
            if (hout32) hout32[(size_t)node * 128 + n] = v;
            atomAddF(&readout[(size_t)g * 384 + n], v);
        }
    }
}

// ---------------------------------------------------------------------------
// Host-side orchestration
// ---------------------------------------------------------------------------

static inline int cdiv_i(long a, long b) { return (int)((a + b - 1) / b); }

extern "C" void kernel_launch(void* const* d_in, const int* in_sizes, int n_in,
                              void* d_out, int out_size, void* d_ws, size_t ws_size,
                              hipStream_t stream) {
    const int*   gate_type = (const int*)d_in[0];
    const int*   srcI      = (const int*)d_in[1];
    const int*   dstI      = (const int*)d_in[2];
    const float* wE        = (const float*)d_in[3];
    const int*   n2g       = (const int*)d_in[4];
    const float* emb       = (const float*)d_in[5];
    const float* W1_0 = (const float*)d_in[6];
    const float* b1_0 = (const float*)d_in[7];
    const float* W2_0 = (const float*)d_in[8];
    const float* W1_1 = (const float*)d_in[9];
    const float* b1_1 = (const float*)d_in[10];
    const float* W2_1 = (const float*)d_in[11];
    const float* W1_2 = (const float*)d_in[12];
    const float* b1_2 = (const float*)d_in[13];
    const float* W2_2 = (const float*)d_in[14];
    (void)n_in; (void)ws_size;

    const int N = in_sizes[0];
    const int E = in_sizes[1];
    const int G = (int)(((long)out_size - (long)N * 128) / 384);

    // Workspace layout
    char* ws = (char*)d_ws;
    const size_t szH = (size_t)N * 128 * sizeof(unsigned short);
    unsigned short* hA = (unsigned short*)(ws);
    unsigned short* hB = (unsigned short*)(ws + szH);
    float*          hN = (float*)(ws + 2 * szH);
    unsigned short* wb = (unsigned short*)(ws + 2 * szH + (size_t)N * 128 * 4);
    unsigned short* WT1_0 = wb;
    unsigned short* WT1_1 = wb + 8192;
    unsigned short* WT1_2 = wb + 24576;
    unsigned short* WT2_0 = wb + 40960;
    unsigned short* WT2_1 = wb + 65536;
    unsigned short* WT2_2 = wb + 98304;

    float* out_h = (float*)d_out;
    float* ro    = out_h + (size_t)N * 128;   // [G][384]

    // Zero graph readout accumulators (re-zeroed every launch: determinism).
    {
        long n = (long)G * 384;
        zero_f32<<<cdiv_i(n, 256), 256, 0, stream>>>(ro, (size_t)n);
    }
    // h0 = bf16(emb[gate_type])
    embed_kernel<<<cdiv_i((long)N * 64, 256), 256, 0, stream>>>(gate_type, emb, hA, N);
    // Weight prep: fp32 row-major -> bf16 col-major
    transpose_bf16_k<<<cdiv_i(64L * 128, 256), 256, 0, stream>>>(W1_0, WT1_0, 64);
    transpose_bf16_k<<<cdiv_i(128L * 128, 256), 256, 0, stream>>>(W1_1, WT1_1, 128);
    transpose_bf16_k<<<cdiv_i(128L * 128, 256), 256, 0, stream>>>(W1_2, WT1_2, 128);
    transpose_bf16_k<<<cdiv_i(192L * 128, 256), 256, 0, stream>>>(W2_0, WT2_0, 192);
    transpose_bf16_k<<<cdiv_i(256L * 128, 256), 256, 0, stream>>>(W2_1, WT2_1, 256);
    transpose_bf16_k<<<cdiv_i(256L * 128, 256), 256, 0, stream>>>(W2_2, WT2_2, 256);

    const int egrid = cdiv_i(E, 128);
    const int ngrid = cdiv_i(N, 128);
    const long hn_elems = (long)N * 128;

    // ----- Layer 0 (Din = 64) -----
    zero_f32<<<2048, 256, 0, stream>>>(hN, (size_t)hn_elems);
    edge_msg_kernel<64><<<egrid, 256, 0, stream>>>(hA, srcI, dstI, wE, W1_0, b1_0,
                                                   WT1_0, hN, E);
    node_update_kernel<64><<<ngrid, 256, 0, stream>>>(hA, hN, WT2_0, n2g, hB,
                                                      nullptr, ro + 0, N);
    // ----- Layer 1 (Din = 128) -----
    zero_f32<<<2048, 256, 0, stream>>>(hN, (size_t)hn_elems);
    edge_msg_kernel<128><<<egrid, 256, 0, stream>>>(hB, srcI, dstI, wE, W1_1, b1_1,
                                                    WT1_1, hN, E);
    node_update_kernel<128><<<ngrid, 256, 0, stream>>>(hB, hN, WT2_1, n2g, hA,
                                                       nullptr, ro + 128, N);
    // ----- Layer 2 (Din = 128), fp32 h -> d_out -----
    zero_f32<<<2048, 256, 0, stream>>>(hN, (size_t)hn_elems);
    edge_msg_kernel<128><<<egrid, 256, 0, stream>>>(hA, srcI, dstI, wE, W1_2, b1_2,
                                                    WT1_2, hN, E);
    node_update_kernel<128><<<ngrid, 256, 0, stream>>>(hA, hN, WT2_2, n2g, hB,
                                                       out_h, ro + 256, N);
}
